// Attention_13116830122301
// MI455X (gfx1250) — matrix operand order repaired and checked by reading, not compiled, not run on hardware
//
#include <hip/hip_runtime.h>

// ---------------------------------------------------------------------------
// Types for CDNA5 WMMA bf16: D(f32 16x16) = A(bf16 16x32) * B(bf16 32x16) + C
// ---------------------------------------------------------------------------
typedef __bf16 bf16_t;
typedef __attribute__((ext_vector_type(8)))  bf16_t v8bf;
typedef __attribute__((ext_vector_type(16))) bf16_t v16bf;
typedef __attribute__((ext_vector_type(8)))  float  v8f;

__device__ __forceinline__ v16bf cat16(v8bf lo, v8bf hi) {
  return __builtin_shufflevector(lo, hi, 0,1,2,3,4,5,6,7,8,9,10,11,12,13,14,15);
}

__device__ __forceinline__ v8f wmma_bf16(v16bf a, v16bf b, v8f c) {
  // (neg_a, A, neg_b, B, c_mod, C, reuse_a, reuse_b)
  return __builtin_amdgcn_wmma_f32_16x16x32_bf16(false, a, false, b, (short)0, c,
                                                 false, false);
}

// Problem constants
#define BATCH 4
#define CH    192
#define C3    576         // 3*CH
#define HEADS 4
#define CPH   48          // CH / HEADS
#define HW    65536       // 256*256
#define IMG_W 256

// ---------------------------------------------------------------------------
// fp32 -> bf16 conversion (grid-stride)
// ---------------------------------------------------------------------------
__global__ void k_cvt_bf16(const float* __restrict__ s, bf16_t* __restrict__ d,
                           size_t n) {
  size_t i = (size_t)blockIdx.x * blockDim.x + threadIdx.x;
  size_t stride = (size_t)gridDim.x * blockDim.x;
  for (; i < n; i += stride) d[i] = (bf16_t)s[i];
}

__global__ void k_zero_f32(float* __restrict__ p, size_t n) {
  size_t i = (size_t)blockIdx.x * blockDim.x + threadIdx.x;
  size_t stride = (size_t)gridDim.x * blockDim.x;
  for (; i < n; i += stride) p[i] = 0.f;
}

// ---------------------------------------------------------------------------
// Tiled WMMA GEMM:  C[M,N] = A[M,K] * B[K,N]   (A,B bf16 row-major)
// Block = 256 threads = 8 waves -> 64x64 output tile.
// Wave layout: waveM in {0,1} (32 rows each), waveN in {0..3} (16 cols each);
// each wave holds two 16x16 accumulators sharing one B fragment.
// B is staged transposed into LDS so both fragments are contiguous b128 reads
// matching the documented 16-bit A(16x32) / B(32x16) VGPR layouts
// (lanes 0-15: K 0-7 / 16-23 for both A and B; lanes 16-31: the other half).
// M % 64 == 0, N % 64 == 0, K % 32 == 0 (holds for 576/192 x 65536 x 192).
// ---------------------------------------------------------------------------
template <typename OUT_T>
__global__ __launch_bounds__(256) void k_gemm_bf16(
    const bf16_t* __restrict__ A, const bf16_t* __restrict__ B,
    OUT_T* __restrict__ C, int N, int K,
    size_t aStride, size_t bStride, size_t cStride) {
  __shared__ __align__(16) bf16_t ldsA[64 * 40];   // [m][k], row stride 40 (80B)
  __shared__ __align__(16) bf16_t ldsBT[64 * 40];  // transposed: [n][k]

  const int tid   = threadIdx.x;
  const int lane  = tid & 31;
  const int wave  = tid >> 5;
  const int waveM = wave & 1;   // 0..1 -> 32 rows each
  const int waveN = wave >> 1;  // 0..3 -> 16 cols each
  const int mBase = blockIdx.y * 64;
  const int nBase = blockIdx.x * 64;

  A += (size_t)blockIdx.z * aStride;
  B += (size_t)blockIdx.z * bStride;
  C += (size_t)blockIdx.z * cStride;

  const int lm     = lane & 15;
  const int kHalfA = (lane < 16) ? 0 : 8;   // A frag K halves
  const int kBaseB = (lane < 16) ? 0 : 16;  // B frag K halves

  // staging coordinates (8 bf16 per thread per tile)
  const int sIdx = tid * 8;
  const int aR = sIdx >> 5, aC = sIdx & 31;   // A: 64 rows x 32 k
  const int bK = sIdx >> 6, bN = sIdx & 63;   // B: 32 k x 64 n

  v8f acc0 = {}, acc1 = {};

  for (int kc = 0; kc < K; kc += 32) {
    // prefetch next K-panel while staging this one
    if (kc + 32 < K) {
      __builtin_prefetch(A + (size_t)(mBase + aR) * K + (kc + 32 + aC), 0, 3);
      __builtin_prefetch(B + (size_t)(kc + 32 + bK) * N + (nBase + bN), 0, 3);
    }
    // Stage A tile 64x32 (one b128 global load + one b128 LDS store / thread)
    {
      v8bf v = *(const v8bf*)(A + (size_t)(mBase + aR) * K + (kc + aC));
      *(v8bf*)&ldsA[aR * 40 + aC] = v;
    }
    // Stage B tile 32x64, transposed into LDS (b128 global load / thread)
    {
      v8bf v = *(const v8bf*)(B + (size_t)(kc + bK) * N + (nBase + bN));
#pragma unroll
      for (int j = 0; j < 8; ++j) ldsBT[(bN + j) * 40 + bK] = v[j];
    }
    __syncthreads();
    {
      int col = waveN * 16 + lm;
      v8bf blo = *(const v8bf*)&ldsBT[col * 40 + kHalfA];
      v8bf bhi = *(const v8bf*)&ldsBT[col * 40 + kHalfA + 16];
      v16bf bf = cat16(blo, bhi);

      int row0 = waveM * 32 + lm;
      v8bf a0lo = *(const v8bf*)&ldsA[row0 * 40 + kHalfA];
      v8bf a0hi = *(const v8bf*)&ldsA[row0 * 40 + kHalfA + 16];
      acc0 = wmma_bf16(cat16(a0lo, a0hi), bf, acc0);

      int row1 = row0 + 16;
      v8bf a1lo = *(const v8bf*)&ldsA[row1 * 40 + kHalfA];
      v8bf a1hi = *(const v8bf*)&ldsA[row1 * 40 + kHalfA + 16];
      acc1 = wmma_bf16(cat16(a1lo, a1hi), bf, acc1);
    }
    __syncthreads();
  }

  // C/D layout: lane -> column lm; VGPR i -> row (0..7 | 8..15)
  const int n = nBase + waveN * 16 + lm;
  const int rsel = (lane < 16) ? 0 : 8;
  int rowBase0 = mBase + waveM * 32 + rsel;
#pragma unroll
  for (int i = 0; i < 8; ++i)
    C[(size_t)(rowBase0 + i) * N + n] = (OUT_T)acc0[i];
  int rowBase1 = rowBase0 + 16;
#pragma unroll
  for (int i = 0; i < 8; ++i)
    C[(size_t)(rowBase1 + i) * N + n] = (OUT_T)acc1[i];
}

// ---------------------------------------------------------------------------
// 3x3 depthwise conv, SAME padding, bf16 in/out + fused per-channel sum(x^2)
// for the q/k L2 norms (channels 0..383). Block = one image row, 256 cols.
// ---------------------------------------------------------------------------
__global__ __launch_bounds__(256) void k_dwconv(
    const bf16_t* __restrict__ in, const float* __restrict__ w,
    bf16_t* __restrict__ out, float* __restrict__ sumsq) {
  const int row = blockIdx.x;
  const int ch  = blockIdx.y;
  const int b   = blockIdx.z;
  const int col = threadIdx.x;
  const size_t img = ((size_t)b * C3 + ch) * HW;

  float w9[9];
#pragma unroll
  for (int i = 0; i < 9; ++i) w9[i] = w[ch * 9 + i];

  float acc = 0.f;
#pragma unroll
  for (int dy = -1; dy <= 1; ++dy) {
    int r = row + dy;
    if ((unsigned)r < (unsigned)IMG_W) {
#pragma unroll
      for (int dx = -1; dx <= 1; ++dx) {
        int c = col + dx;
        if ((unsigned)c < (unsigned)IMG_W)
          acc += w9[(dy + 1) * 3 + (dx + 1)] * (float)in[img + r * IMG_W + c];
      }
    }
  }
  out[img + row * IMG_W + col] = (bf16_t)acc;

  if (ch < 2 * CH) {  // q and k channels need row norms
    __shared__ float red[256];
    red[col] = acc * acc;
    __syncthreads();
    for (int s = 128; s > 0; s >>= 1) {
      if (col < s) red[col] += red[col + s];
      __syncthreads();
    }
    if (col == 0) atomicAdd(&sumsq[b * (2 * CH) + ch], red[0]);
  }
}

// ---------------------------------------------------------------------------
// Gram matrices G[b,h] = q[b,h] (48 x HW) * k[b,h]^T  via WMMA bf16.
// K-dim (pixels) is contiguous for BOTH operands -> direct aligned b128
// global loads, no LDS. Partial 48x48 results accumulate with atomicAdd.
// One wave per block; 3x3 grid of 16x16 accumulators.
// ---------------------------------------------------------------------------
__global__ __launch_bounds__(32) void k_gram(
    const bf16_t* __restrict__ qkv, float* __restrict__ S, int chunk) {
  const int bh = blockIdx.y;  // b*HEADS + h
  const int b = bh >> 2, h = bh & 3;
  const int lane = threadIdx.x;
  const int lm     = lane & 15;
  const int kHalfA = (lane < 16) ? 0 : 8;
  const int kBaseB = (lane < 16) ? 0 : 16;

  const bf16_t* qb = qkv + ((size_t)b * C3 + h * CPH) * HW;
  const bf16_t* kb = qkv + ((size_t)b * C3 + CH + h * CPH) * HW;
  const int n0 = blockIdx.x * chunk;

  v8f acc[3][3];
#pragma unroll
  for (int i = 0; i < 3; ++i)
#pragma unroll
    for (int j = 0; j < 3; ++j) acc[i][j] = (v8f){};

  for (int n = n0; n < n0 + chunk; n += 32) {
    v16bf af[3], bfr[3];
#pragma unroll
    for (int t = 0; t < 3; ++t) {
      const bf16_t* pq = qb + (size_t)(t * 16 + lm) * HW + n + kHalfA;
      af[t] = cat16(*(const v8bf*)pq, *(const v8bf*)(pq + 16));
      const bf16_t* pk = kb + (size_t)(t * 16 + lm) * HW + n + kHalfA;
      bfr[t] = cat16(*(const v8bf*)pk, *(const v8bf*)(pk + 16));
    }
#pragma unroll
    for (int mt = 0; mt < 3; ++mt)
#pragma unroll
      for (int nt = 0; nt < 3; ++nt)
        acc[mt][nt] = wmma_bf16(af[mt], bfr[nt], acc[mt][nt]);
  }

  float* Sb = S + (size_t)bh * CPH * CPH;
  const int rb = (lane < 16) ? 0 : 8;
#pragma unroll
  for (int mt = 0; mt < 3; ++mt)
#pragma unroll
    for (int nt = 0; nt < 3; ++nt)
#pragma unroll
      for (int i = 0; i < 8; ++i)
        atomicAdd(&Sb[(mt * 16 + rb + i) * CPH + nt * 16 + lm], acc[mt][nt][i]);
}

// ---------------------------------------------------------------------------
// Softmax over normalized Gram rows:
// attn[c,d] = softmax_d( G[c,d] / (max(|q_c|,eps)*max(|k_d|,eps)) * T[h] )
// ---------------------------------------------------------------------------
__global__ __launch_bounds__(64) void k_softmax(
    const float* __restrict__ S, const float* __restrict__ sumsq,
    const float* __restrict__ temp, float* __restrict__ attn) {
  const int bh = blockIdx.x;
  const int b = bh >> 2, h = bh & 3;
  __shared__ float nk[CPH];
  const int t = threadIdx.x;
  if (t < CPH)
    nk[t] = fmaxf(sqrtf(sumsq[b * (2 * CH) + CH + h * CPH + t]), 1e-12f);
  __syncthreads();
  if (t < CPH) {
    float nq = fmaxf(sqrtf(sumsq[b * (2 * CH) + h * CPH + t]), 1e-12f);
    float tm = temp[h];
    const float* Sr = S + ((size_t)bh * CPH + t) * CPH;
    float lg[CPH], mx = -1e30f;
    for (int d = 0; d < CPH; ++d) {
      lg[d] = Sr[d] / (nq * nk[d]) * tm;
      mx = fmaxf(mx, lg[d]);
    }
    float sum = 0.f;
    for (int d = 0; d < CPH; ++d) { lg[d] = __expf(lg[d] - mx); sum += lg[d]; }
    float inv = 1.f / sum;
    float* Ar = attn + ((size_t)bh * CPH + t) * CPH;
    for (int d = 0; d < CPH; ++d) Ar[d] = lg[d] * inv;
  }
}

// ---------------------------------------------------------------------------
// Fold softmax + projection:  Wb[b][o,dg] = sum_c proj_w[o, h*48+c]*attn[b,h,c,d]
// so the final output is a single GEMM  out[b] = Wb[b] @ v[b].
// ---------------------------------------------------------------------------
__global__ __launch_bounds__(192) void k_build_wb(
    const float* __restrict__ proj_w, const float* __restrict__ attn,
    bf16_t* __restrict__ Wb) {
  const int bo = blockIdx.x;  // b*192 + o
  const int b = bo / CH, o = bo % CH;
  const int dg = threadIdx.x;  // 0..191
  const int h = dg / CPH, d = dg % CPH;
  const float* pw = proj_w + (size_t)o * CH + h * CPH;
  const float* at = attn + ((size_t)(b * HEADS + h) * CPH) * CPH + d;
  float s = 0.f;
#pragma unroll 8
  for (int c = 0; c < CPH; ++c) s += pw[c] * at[c * CPH];
  Wb[((size_t)b * CH + o) * CH + dg] = (bf16_t)s;
}

// ---------------------------------------------------------------------------
// Launch
// ---------------------------------------------------------------------------
extern "C" void kernel_launch(void* const* d_in, const int* in_sizes, int n_in,
                              void* d_out, int out_size, void* d_ws,
                              size_t ws_size, hipStream_t stream) {
  const float* x      = (const float*)d_in[0];
  const float* qkv_w  = (const float*)d_in[1];
  const float* dw_w   = (const float*)d_in[2];
  const float* proj_w = (const float*)d_in[3];
  const float* temp   = (const float*)d_in[4];

  char* ws = (char*)d_ws;
  size_t off = 0;
  auto carve = [&](size_t bytes) -> char* {
    char* p = ws + off;
    off += (bytes + 255) & ~(size_t)255;
    return p;
  };

  const size_t nX   = (size_t)BATCH * CH * HW;     // 50.3M
  const size_t nQKV = (size_t)BATCH * C3 * HW;     // 151M

  bf16_t* xb    = (bf16_t*)carve(nX * sizeof(bf16_t));
  bf16_t* wq    = (bf16_t*)carve((size_t)C3 * CH * sizeof(bf16_t));
  bf16_t* qlin  = (bf16_t*)carve(nQKV * sizeof(bf16_t));
  bf16_t* qdw   = (bf16_t*)carve(nQKV * sizeof(bf16_t));
  float*  sumsq = (float*)carve((size_t)BATCH * 2 * CH * sizeof(float));
  float*  S     = (float*)carve((size_t)BATCH * HEADS * CPH * CPH * sizeof(float));
  float*  attn  = (float*)carve((size_t)BATCH * HEADS * CPH * CPH * sizeof(float));
  bf16_t* Wb    = (bf16_t*)carve((size_t)BATCH * CH * CH * sizeof(bf16_t));

  // Precision staging
  k_cvt_bf16<<<2048, 256, 0, stream>>>(x, xb, nX);
  k_cvt_bf16<<<128, 256, 0, stream>>>(qkv_w, wq, (size_t)C3 * CH);
  k_zero_f32<<<8, 256, 0, stream>>>(sumsq, (size_t)BATCH * 2 * CH);
  k_zero_f32<<<64, 256, 0, stream>>>(S, (size_t)BATCH * HEADS * CPH * CPH);

  // qkv = qkv_w @ x   (per batch: 576 x 65536, K=192)
  k_gemm_bf16<bf16_t><<<dim3(HW / 64, C3 / 64, BATCH), 256, 0, stream>>>(
      wq, xb, qlin, HW, CH, /*aStride=*/0, (size_t)CH * HW, (size_t)C3 * HW);

  // depthwise 3x3 + fused q/k sum-of-squares
  k_dwconv<<<dim3(IMG_W, C3, BATCH), 256, 0, stream>>>(qlin, dw_w, qdw, sumsq);

  // Gram matrices q k^T per (b, head)
  k_gram<<<dim3(256, BATCH * HEADS), 32, 0, stream>>>(qdw, S, HW / 256);

  // normalize + softmax
  k_softmax<<<BATCH * HEADS, 64, 0, stream>>>(S, sumsq, temp, attn);

  // fold projection through attention
  k_build_wb<<<BATCH * CH, 192, 0, stream>>>(proj_w, attn, Wb);

  // out[b] = Wb[b] @ v[b]   (192 x 65536, K=192), fp32 out
  k_gemm_bf16<float><<<dim3(HW / 64, CH / 64, BATCH), 256, 0, stream>>>(
      Wb, qdw + (size_t)2 * CH * HW, (float*)d_out, HW, CH,
      (size_t)CH * CH, (size_t)C3 * HW, (size_t)CH * HW);
}
